// AdvancedLSTMModel_49031346651542
// MI455X (gfx1250) — matrix operand order repaired and checked by reading, not compiled, not run on hardware
//
#include <hip/hip_runtime.h>

// ---------------------------------------------------------------------------
// AdvancedLSTMModel on MI455X (gfx1250), wave32 + WMMA f16->f32.
// Seq len == 1 so attention is identity; model is two fused GEMM+LSTM-cell
// stages plus a dot product. f-gate is dead (c_prev==0) and skipped.
// ---------------------------------------------------------------------------

typedef __attribute__((ext_vector_type(16))) _Float16 v16h;
typedef __attribute__((ext_vector_type(8)))  _Float16 v8h;
typedef __attribute__((ext_vector_type(8)))  float    v8f;
typedef __attribute__((ext_vector_type(4)))  float    v4f;

#define NB 262144          // batch
#define DD 512             // input dim
#define NT 6               // live gate n-tiles: {i,g,o} x {fwd,bwd}
#define W0FRAG_HALVES (16*NT*512)      // 49152 halves = 96 KB
#define W1FRAG_HALVES (NT*512)         // 3072 halves  = 6 KB
#define HPAD 40                        // h-scratch row stride (halves) -> 80B, 16B aligned
#define SMEM_BYTES ((W0FRAG_HALVES + W1FRAG_HALVES)*2 + 8*16*HPAD*2)  // 114688

__device__ __forceinline__ float fsig(float x)  { return 1.0f / (1.0f + __expf(-x)); }
__device__ __forceinline__ float ftanh(float x) { return 1.0f - 2.0f / (1.0f + __expf(2.0f * x)); }

// live n-tile index -> column tile of the 128-wide gate matrix
__device__ __forceinline__ int ntOf(int cn) { return cn + (cn >= 1) + (cn >= 4); }

// ---------------------------------------------------------------------------
// Kernel 0: pre-swizzle weights into WMMA B-fragment layout (f16), f-gate
// tiles dropped. B 32x16 f16 layout: n = lane%16, e -> k = 16*(e/8) + 8*(lane/16) + e%8.
// Fragment order: frag = ks*NT + cn  (one k-step's 6 fragments span 6 KB so
// all DS loads encode as immediate offsets off one base register).
// Per-fragment storage: [chunk e<8 | chunk e>=8], each chunk = 32 lanes x 16B
// so fragment loads are two ds_load_b128 with 16B lane stride (2 acc/bank).
// flat half index within fragment: (e/8)*256 + lane*8 + (e%8)
// ---------------------------------------------------------------------------
__global__ void lstm_prep_weights(const float* __restrict__ wf0,
                                  const float* __restrict__ wb0,
                                  const float* __restrict__ wf1,
                                  const float* __restrict__ wb1,
                                  _Float16* __restrict__ w0frag,
                                  _Float16* __restrict__ w1frag) {
  int idx = blockIdx.x * 256 + threadIdx.x;
  if (idx < W0FRAG_HALVES) {
    int f = idx >> 9, r = idx & 511;
    int chunk = r >> 8, t = r & 255;
    int lane = t >> 3, e = chunk * 8 + (t & 7);
    int ks = f / NT, cn = f % NT;       // frag = ks*NT + cn
    int row = ntOf(cn) * 16 + (lane & 15);        // gate column (0..127)
    int col = ks * 32 + chunk * 16 + (lane >> 4) * 8 + (t & 7);  // k (0..511)
    float v = (row < 64) ? wf0[row * DD + col] : wb0[(row - 64) * DD + col];
    w0frag[idx] = (_Float16)v;
  } else if (idx < W0FRAG_HALVES + W1FRAG_HALVES) {
    int j = idx - W0FRAG_HALVES;
    int f = j >> 9, r = j & 511;
    int chunk = r >> 8, t = r & 255;
    int lane = t >> 3, e = chunk * 8 + (t & 7);
    int row = ntOf(f) * 16 + (lane & 15);
    int col = chunk * 16 + (lane >> 4) * 8 + (t & 7);  // k (0..31)
    float v = (row < 64) ? wf1[row * 32 + col] : wb1[(row - 64) * 32 + col];
    w1frag[j] = (_Float16)v;
  }
}

// ---------------------------------------------------------------------------
// Main kernel: 256 threads = 8 waves; each wave owns a 16-row tile of x.
// ---------------------------------------------------------------------------
__global__ __launch_bounds__(256)
void lstm_main(const float* __restrict__ x,
               const float* __restrict__ bf0, const float* __restrict__ bb0,
               const float* __restrict__ bf1, const float* __restrict__ bb1,
               const float* __restrict__ fcw, const float* __restrict__ fcb,
               const _Float16* __restrict__ wfrag,   // w0frag ++ w1frag
               float* __restrict__ out) {
  extern __shared__ __align__(16) unsigned char smem_raw[];
  _Float16* sW0 = (_Float16*)smem_raw;
  _Float16* sW1 = sW0 + W0FRAG_HALVES;

  const int tid = threadIdx.x;

  // Stage swizzled weight fragments (102 KB) into LDS, 16B vector copies.
  {
    float4*       dst = (float4*)smem_raw;
    const float4* src = (const float4*)wfrag;
    const int n16 = (W0FRAG_HALVES + W1FRAG_HALVES) * 2 / 16;  // 6528
    for (int i = tid; i < n16; i += 256) dst[i] = src[i];
  }
  __syncthreads();

  const int wave = tid >> 5;
  const int lane = tid & 31;
  const int g    = lane >> 4;     // lane half
  const int c15  = lane & 15;
  const int m    = c15;           // A-matrix row within tile
  const int rowBase = blockIdx.x * 128 + wave * 16;

  _Float16* sH = sW0 + (W0FRAG_HALVES + W1FRAG_HALVES) + wave * 16 * HPAD;

  // Per-lane gate biases (column = nt*16 + c15) and fc weights.
  float b0v[NT], b1v[NT];
  {
    const int off[NT] = {0, 32, 48, 0, 32, 48};   // i,g,o offsets per direction
#pragma unroll
    for (int cn = 0; cn < NT; cn++) {
      b0v[cn] = (cn < 3 ? bf0 : bb0)[off[cn] + c15];
      b1v[cn] = (cn < 3 ? bf1 : bb1)[off[cn] + c15];
    }
  }
  const float fwlo = fcw[c15], fwhi = fcw[16 + c15], fcbias = fcb[0];

  // A-stream base for this lane: row m of the tile, lane-half k offset g*8.
  const float* xrow = x + (size_t)(rowBase + m) * DD + g * 8;

  // ---------------- Stage 1: gates0 = x @ W0cat.T (6 live n-tiles) ---------
  v8f acc[NT];
#pragma unroll
  for (int cn = 0; cn < NT; cn++) acc[cn] = (v8f){0, 0, 0, 0, 0, 0, 0, 0};

  for (int ks = 0; ks < 16; ks++) {
    // Issue the global A loads first (loadcnt), ...
    const v4f* pa = (const v4f*)(xrow + ks * 32);
    v4f r0 = pa[0], r1 = pa[1];      // k = g*8 .. g*8+7
    v4f r2 = pa[4], r3 = pa[5];      // k = 16+g*8 .. 16+g*8+7
    if (ks < 15) __builtin_prefetch(xrow + (ks + 1) * 32, 0, 1);

    // ... then ALL 12 B-fragment DS loads (dscnt, 6 KB window, immediate
    // offsets off one base), THEN the WMMA chain -> partial waits + MLP.
    const _Float16* fb = sW0 + ks * (NT * 512) + lane * 8;
    v8h blo[NT], bhi[NT];
#pragma unroll
    for (int cn = 0; cn < NT; cn++) {
      blo[cn] = *(const v8h*)(fb + cn * 512);
      bhi[cn] = *(const v8h*)(fb + cn * 512 + 256);
    }

    v16h A;
#pragma unroll
    for (int i = 0; i < 4; i++) {
      A[i]      = (_Float16)r0[i];
      A[4 + i]  = (_Float16)r1[i];
      A[8 + i]  = (_Float16)r2[i];
      A[12 + i] = (_Float16)r3[i];
    }

#pragma unroll
    for (int cn = 0; cn < NT; cn++) {
      v16h Bf;
#pragma unroll
      for (int i = 0; i < 8; i++) { Bf[i] = blo[cn][i]; Bf[8 + i] = bhi[cn][i]; }
      acc[cn] = __builtin_amdgcn_wmma_f32_16x16x32_f16(
          false, A, false, Bf, (short)0, acc[cn], false, false);
    }
  }

  // LSTM cell 1 (f gate dead: c = i*g); write h0 tile (16x32 f16) to LDS.
#pragma unroll
  for (int j = 0; j < 8; j++) {
    float i_f = fsig (acc[0][j] + b0v[0]);
    float g_f = ftanh(acc[1][j] + b0v[1]);
    float o_f = fsig (acc[2][j] + b0v[2]);
    float hf  = o_f * ftanh(i_f * g_f);
    float i_b = fsig (acc[3][j] + b0v[3]);
    float g_b = ftanh(acc[4][j] + b0v[4]);
    float o_b = fsig (acc[5][j] + b0v[5]);
    float hb  = o_b * ftanh(i_b * g_b);
    int row = g * 8 + j;                       // C-layout row
    sH[row * HPAD + c15]      = (_Float16)hf;  // fwd -> cols 0..15
    sH[row * HPAD + 16 + c15] = (_Float16)hb;  // bwd -> cols 16..31
  }
  // Same-wave DS ops are in order (DScnt); private region -> no barrier.

  // ---------------- Stage 2: gates1 = h0 @ W1cat.T (K=32, one k-step) ------
  v16h A1;
  {
    const _Float16* hr = sH + m * HPAD;
    v8h alo = *(const v8h*)(hr + g * 8);        // k = g*8 .. g*8+7
    v8h ahi = *(const v8h*)(hr + 16 + g * 8);   // k = 16+g*8 ..
#pragma unroll
    for (int i = 0; i < 8; i++) { A1[i] = alo[i]; A1[8 + i] = ahi[i]; }
  }

  v8h blo1[NT], bhi1[NT];
#pragma unroll
  for (int cn = 0; cn < NT; cn++) {
    blo1[cn] = *(const v8h*)(sW1 + cn * 512 + lane * 8);
    bhi1[cn] = *(const v8h*)(sW1 + cn * 512 + 256 + lane * 8);
  }

  v8f acc1[NT];
#pragma unroll
  for (int cn = 0; cn < NT; cn++) {
    v16h Bf;
#pragma unroll
    for (int i = 0; i < 8; i++) { Bf[i] = blo1[cn][i]; Bf[8 + i] = bhi1[cn][i]; }
    v8f z = (v8f){0, 0, 0, 0, 0, 0, 0, 0};
    acc1[cn] = __builtin_amdgcn_wmma_f32_16x16x32_f16(
        false, A1, false, Bf, (short)0, z, false, false);
  }

  // LSTM cell 2 + fc dot product (attention over len-1 seq == identity).
#pragma unroll
  for (int j = 0; j < 8; j++) {
    float i_f = fsig (acc1[0][j] + b1v[0]);
    float g_f = ftanh(acc1[1][j] + b1v[1]);
    float o_f = fsig (acc1[2][j] + b1v[2]);
    float h1f = o_f * ftanh(i_f * g_f);
    float i_b = fsig (acc1[3][j] + b1v[3]);
    float g_b = ftanh(acc1[4][j] + b1v[4]);
    float o_b = fsig (acc1[5][j] + b1v[5]);
    float h1b = o_b * ftanh(i_b * g_b);

    float part = h1f * fwlo + h1b * fwhi;      // per-column partial
    part += __shfl_xor(part, 1, 32);           // reduce over 16 columns
    part += __shfl_xor(part, 2, 32);
    part += __shfl_xor(part, 4, 32);
    part += __shfl_xor(part, 8, 32);
    if (c15 == 0) out[rowBase + g * 8 + j] = part + fcbias;
  }
}

// ---------------------------------------------------------------------------
extern "C" void kernel_launch(void* const* d_in, const int* in_sizes, int n_in,
                              void* d_out, int out_size, void* d_ws, size_t ws_size,
                              hipStream_t stream) {
  const float* x   = (const float*)d_in[0];
  const float* wf0 = (const float*)d_in[1];
  const float* bf0 = (const float*)d_in[2];
  const float* wb0 = (const float*)d_in[3];
  const float* bb0 = (const float*)d_in[4];
  const float* wf1 = (const float*)d_in[5];
  const float* bf1 = (const float*)d_in[6];
  const float* wb1 = (const float*)d_in[7];
  const float* bb1 = (const float*)d_in[8];
  // d_in[9]=attn_w, d_in[10]=attn_b: mathematically dead (softmax over len 1).
  const float* fcw = (const float*)d_in[11];
  const float* fcb = (const float*)d_in[12];
  float* out = (float*)d_out;

  _Float16* w0frag = (_Float16*)d_ws;
  _Float16* w1frag = w0frag + W0FRAG_HALVES;

  // 52224 swizzle elements / 256 threads = 204 blocks exactly.
  lstm_prep_weights<<<204, 256, 0, stream>>>(wf0, wb0, wf1, wb1, w0frag, w1frag);

  (void)hipFuncSetAttribute((const void*)lstm_main,
                            hipFuncAttributeMaxDynamicSharedMemorySize,
                            SMEM_BYTES);
  lstm_main<<<NB / 128, 256, SMEM_BYTES, stream>>>(
      x, bf0, bb0, bf1, bb1, fcw, fcb, w0frag, out);
}
